// MultiHeadSelfAttention_88098369176309
// MI455X (gfx1250) — compile-verified, matrix operand order checked by reading, not run on hardware
//
#include <hip/hip_runtime.h>

#define B_  2
#define T_  4096
#define D_  512
#define H_  8
#define HD_ 64
#define NELEM ((size_t)B_ * T_ * D_)   // 4,194,304 elements
#define WELEM ((size_t)D_ * D_)        // 262,144 elements

typedef __bf16 bf16_t;
typedef bf16_t bf16x16 __attribute__((ext_vector_type(16)));
typedef bf16_t bf16x2  __attribute__((ext_vector_type(2)));
typedef float  f32x8   __attribute__((ext_vector_type(8)));

// ---- async global->LDS probe (graceful fallback if builtins absent) -------
#if defined(__HIP_DEVICE_COMPILE__) && defined(__gfx1250__) && defined(__has_builtin)
#  if __has_builtin(__builtin_amdgcn_global_load_async_to_lds_b128) && \
      __has_builtin(__builtin_amdgcn_s_wait_asynccnt)
#    define USE_ASYNC 1
#  endif
#endif
#ifndef USE_ASYNC
#define USE_ASYNC 0
#endif

// builtin expects int4-vector pointers (from compiler diagnostic)
typedef int i32x4_t __attribute__((vector_size(16)));
typedef __attribute__((address_space(1))) i32x4_t as1_i32x4;
typedef __attribute__((address_space(3))) i32x4_t as3_i32x4;

// copy 16B global -> LDS (per lane)
__device__ __forceinline__ void cp_b128(const unsigned short* g, unsigned short* l) {
#if USE_ASYNC
    __builtin_amdgcn_global_load_async_to_lds_b128((as1_i32x4*)g, (as3_i32x4*)l, 0, 0);
#else
    *(uint4*)l = *(const uint4*)g;
#endif
}

__device__ __forceinline__ void cp_wait() {
#if USE_ASYNC
    __builtin_amdgcn_s_wait_asynccnt(0);
#endif
}

// ---------------------------------------------------------------------------
__device__ __forceinline__ unsigned short f2bfu(float f) {
    unsigned u = __builtin_bit_cast(unsigned, f);
    u += 0x7FFFu + ((u >> 16) & 1u);   // round-to-nearest-even
    return (unsigned short)(u >> 16);
}

__device__ __forceinline__ f32x8 zero8() {
    f32x8 z = {0.f, 0.f, 0.f, 0.f, 0.f, 0.f, 0.f, 0.f};
    return z;
}

// A/B 16-bit fragment, k contiguous within one LDS row.
// vgpr v holds k pair {base,base+1}; base = (v<4 ? 2v : 2v+8) + 8*half.
// Contiguous k-pairs let the compiler merge these into 2x ds_load_b128.
__device__ __forceinline__ bf16x16 frag_krow(const unsigned short* rowptr, int h) {
    bf16x16 f;
#pragma unroll
    for (int v = 0; v < 8; ++v) {
        int k = ((v < 4) ? (2 * v) : (2 * v + 8)) + 8 * h;
        unsigned w = *(const unsigned*)(rowptr + k);
        bf16x2 t = __builtin_bit_cast(bf16x2, w);
        f[2 * v]     = t[0];
        f[2 * v + 1] = t[1];
    }
    return f;
}

__device__ __forceinline__ f32x8 wmma_bf16(bf16x16 a, bf16x16 b, f32x8 c) {
    return __builtin_amdgcn_wmma_f32_16x16x32_bf16(false, a, false, b, (short)0, c, false, false);
}

// ---------------------------------------------------------------------------
// Prep 1: X fp32 -> bf16, one pass (8 elements / thread, b128 in/out)
// ---------------------------------------------------------------------------
__global__ void x_to_bf16_kernel(const float* __restrict__ in,
                                 unsigned short* __restrict__ out) {
    size_t i = (size_t)blockIdx.x * blockDim.x + threadIdx.x;
    const float4* src = (const float4*)in + 2 * i;
    float4 a = src[0], b = src[1];
    uint4 r;
    r.x = (unsigned)f2bfu(a.x) | ((unsigned)f2bfu(a.y) << 16);
    r.y = (unsigned)f2bfu(a.z) | ((unsigned)f2bfu(a.w) << 16);
    r.z = (unsigned)f2bfu(b.x) | ((unsigned)f2bfu(b.y) << 16);
    r.w = (unsigned)f2bfu(b.z) | ((unsigned)f2bfu(b.w) << 16);
    ((uint4*)out)[i] = r;
}

// ---------------------------------------------------------------------------
// Prep 2: W [in][out] fp32 -> Wt [out][in] bf16 for all 4 weight matrices.
// grid (16,16,4), block (32,8); 32x32 tiles via LDS.
// ---------------------------------------------------------------------------
__global__ void w_transpose_kernel(const float* __restrict__ WK, const float* __restrict__ WQ,
                                   const float* __restrict__ WV, const float* __restrict__ WO,
                                   unsigned short* __restrict__ WtK, unsigned short* __restrict__ WtQ,
                                   unsigned short* __restrict__ WtV, unsigned short* __restrict__ WtO) {
    __shared__ unsigned short tile[32][33];
    const int z = blockIdx.z;
    const float* W = (z == 0) ? WK : (z == 1) ? WQ : (z == 2) ? WV : WO;
    unsigned short* Wt = (z == 0) ? WtK : (z == 1) ? WtQ : (z == 2) ? WtV : WtO;
    const int j0 = blockIdx.x * 32, i0 = blockIdx.y * 32;
    const int tx = threadIdx.x, ty = threadIdx.y;
#pragma unroll
    for (int r = 0; r < 4; ++r) {
        int i = i0 + ty + 8 * r;
        tile[ty + 8 * r][tx] = f2bfu(W[(size_t)i * D_ + j0 + tx]);
    }
    __syncthreads();
#pragma unroll
    for (int r = 0; r < 4; ++r) {
        int o = j0 + ty + 8 * r;
        Wt[(size_t)o * D_ + i0 + tx] = tile[tx][ty + 8 * r];
    }
}

// ---------------------------------------------------------------------------
// Kernel 1: fused QKV projection, 16x64 output tile per wave.
// Q,K written head-split [B,H,T,hd]; V written transposed [B,H,hd,T].
// grid = (D/64, (B*T)/16, 3), block = 32
// ---------------------------------------------------------------------------
__global__ void qkv_proj_kernel(const unsigned short* __restrict__ Xb,
                                const unsigned short* __restrict__ WtK, const float* __restrict__ bKp,
                                const unsigned short* __restrict__ WtQ, const float* __restrict__ bQp,
                                const unsigned short* __restrict__ WtV, const float* __restrict__ bVp,
                                unsigned short* __restrict__ Qb,
                                unsigned short* __restrict__ Kb,
                                unsigned short* __restrict__ Vt) {
    __shared__ __align__(16) unsigned short Xs[16][32];
    __shared__ __align__(16) unsigned short Ws[64][32];

    const int lane = threadIdx.x;
    const int n = lane & 15, h = lane >> 4;
    const int ct = blockIdx.x;      // 64-wide column tile
    const int ry = blockIdx.y;      // 16-row tile
    const int which = blockIdx.z;   // 0=K 1=Q 2=V

    const unsigned short* Wt = (which == 0) ? WtK : (which == 1) ? WtQ : WtV;
    const float* bias = (which == 0) ? bKp : (which == 1) ? bQp : bVp;

    f32x8 acc[4] = {zero8(), zero8(), zero8(), zero8()};

    for (int kk = 0; kk < D_; kk += 32) {
        __syncthreads();
        {   // X tile: lane -> row n, 16 bf16 at [kk + 16h]
            const unsigned short* xs = Xb + (size_t)(ry * 16 + n) * D_ + kk + 16 * h;
            cp_b128(xs,     &Xs[n][16 * h]);
            cp_b128(xs + 8, &Xs[n][16 * h + 8]);
        }
        {   // Wt tile: 64 out-rows x 32 k; lane covers rows {lane, lane+32}
#pragma unroll
            for (int r = 0; r < 2; ++r) {
                int wr = lane + 32 * r;
                const unsigned short* src = Wt + (size_t)(64 * ct + wr) * D_ + kk;
#pragma unroll
                for (int i = 0; i < 4; ++i) cp_b128(src + 8 * i, &Ws[wr][8 * i]);
            }
        }
        cp_wait();
        __syncthreads();
        bf16x16 a = frag_krow(&Xs[n][0], h);
#pragma unroll
        for (int i = 0; i < 4; ++i)
            acc[i] = wmma_bf16(a, frag_krow(&Ws[16 * i + n][0], h), acc[i]);
    }

    const int b  = (ry * 16) >> 12;         // / T
    const int t0 = (ry * 16) & (T_ - 1);
#pragma unroll
    for (int i = 0; i < 4; ++i) {
        int col = 64 * ct + 16 * i + n;
        float bv = bias[col];
        int head = col >> 6, d = col & (HD_ - 1);
        if (which != 2) {
            unsigned short* op = ((which == 0) ? Kb : Qb) +
                (((size_t)b * H_ + head) * T_ + t0 + 8 * h) * HD_ + d;
#pragma unroll
            for (int v = 0; v < 8; ++v) op[v * HD_] = f2bfu(acc[i][v] + bv);
        } else {
            unsigned short* op = Vt +
                (((size_t)b * H_ + head) * HD_ + d) * T_ + t0 + 8 * h;
#pragma unroll
            for (int v = 0; v < 8; ++v) op[v] = f2bfu(acc[i][v] + bv);
        }
    }
}

// ---------------------------------------------------------------------------
// Kernel 2: causal flash attention, online softmax, 32 keys / step.
// V consumed from transposed layout so every fragment is k-contiguous.
// grid = (T/16, B*H), block = 32
// ---------------------------------------------------------------------------
__global__ void flash_attn_kernel(const unsigned short* __restrict__ Qb,
                                  const unsigned short* __restrict__ Kb,
                                  const unsigned short* __restrict__ Vt,
                                  unsigned short* __restrict__ Zb) {
    __shared__ __align__(16) unsigned short Qs[16][64];
    __shared__ __align__(16) unsigned short Ks[32][64];
    __shared__ __align__(16) unsigned short Vts[64][32];
    __shared__ __align__(16) unsigned short Ps[16][32];

    const int lane = threadIdx.x;
    const int n = lane & 15, h = lane >> 4;
    const int qi = blockIdx.x;
    const int bh = blockIdx.y;
    const int b = bh >> 3, head = bh & (H_ - 1);

    const unsigned short* Qbh = Qb + (size_t)bh * T_ * HD_;
    const unsigned short* Kbh = Kb + (size_t)bh * T_ * HD_;
    const unsigned short* Vbh = Vt + (size_t)bh * HD_ * T_;

    {   // Q tile 16x64: lane -> row n, 32 bf16 at 32h
        const unsigned short* src = Qbh + (size_t)(16 * qi + n) * HD_ + 32 * h;
#pragma unroll
        for (int i = 0; i < 4; ++i) cp_b128(src + 8 * i, &Qs[n][32 * h + 8 * i]);
        cp_wait();
    }
    __syncthreads();
    const bf16x16 aq0 = frag_krow(&Qs[n][0], h);
    const bf16x16 aq1 = frag_krow(&Qs[n][32], h);

    f32x8 o[4] = {zero8(), zero8(), zero8(), zero8()};
    float mrow[8], lrow[8];
#pragma unroll
    for (int v = 0; v < 8; ++v) { mrow[v] = -1e30f; lrow[v] = 0.f; }

    const int nsteps = (qi + 2) >> 1;
    const float scale = 0.125f;     // 1/sqrt(64)

    for (int j = 0; j < nsteps; ++j) {
        __syncthreads();
        {   // K tile 32x64: lane -> one key row (8x b128)
            const unsigned short* ks = Kbh + (size_t)(32 * j + lane) * HD_;
#pragma unroll
            for (int i = 0; i < 8; ++i) cp_b128(ks + 8 * i, &Ks[lane][8 * i]);
            // Vt tile 64x32: lane -> d rows {lane, lane+32} (4x b128 each)
#pragma unroll
            for (int r = 0; r < 2; ++r) {
                int d = lane + 32 * r;
                const unsigned short* vs = Vbh + (size_t)d * T_ + 32 * j;
#pragma unroll
                for (int i = 0; i < 4; ++i) cp_b128(vs + 8 * i, &Vts[d][8 * i]);
            }
            cp_wait();
        }
        __syncthreads();

        // S = Q (16x64) @ K^T (64x32)
        f32x8 s0 = zero8(), s1 = zero8();
        s0 = wmma_bf16(aq0, frag_krow(&Ks[n][0],       h), s0);
        s0 = wmma_bf16(aq1, frag_krow(&Ks[n][32],      h), s0);
        s1 = wmma_bf16(aq0, frag_krow(&Ks[16 + n][0],  h), s1);
        s1 = wmma_bf16(aq1, frag_krow(&Ks[16 + n][32], h), s1);

        float p0[8], p1[8], fac[8];
#pragma unroll
        for (int v = 0; v < 8; ++v) {
            int m   = v + 8 * h;
            int row = 16 * qi + m;
            int c0  = 32 * j + n;
            float x0 = (c0      <= row) ? s0[v] * scale : -1e30f;
            float x1 = (c0 + 16 <= row) ? s1[v] * scale : -1e30f;
            float mx = fmaxf(x0, x1);
            mx = fmaxf(mx, __shfl_xor(mx, 1));
            mx = fmaxf(mx, __shfl_xor(mx, 2));
            mx = fmaxf(mx, __shfl_xor(mx, 4));
            mx = fmaxf(mx, __shfl_xor(mx, 8));
            float mnew = fmaxf(mrow[v], mx);
            float f  = __expf(mrow[v] - mnew);
            float e0 = __expf(x0 - mnew);
            float e1 = __expf(x1 - mnew);
            float sum = e0 + e1;
            sum += __shfl_xor(sum, 1);
            sum += __shfl_xor(sum, 2);
            sum += __shfl_xor(sum, 4);
            sum += __shfl_xor(sum, 8);
            lrow[v] = lrow[v] * f + sum;
            mrow[v] = mnew;
            fac[v] = f;
            p0[v] = e0;
            p1[v] = e1;
        }
#pragma unroll
        for (int t = 0; t < 4; ++t)
#pragma unroll
            for (int v = 0; v < 8; ++v) o[t][v] *= fac[v];

#pragma unroll
        for (int v = 0; v < 8; ++v) {
            int m = v + 8 * h;
            Ps[m][n]      = f2bfu(p0[v]);
            Ps[m][16 + n] = f2bfu(p1[v]);
        }
        __syncthreads();
        const bf16x16 ap = frag_krow(&Ps[n][0], h);
#pragma unroll
        for (int t = 0; t < 4; ++t)
            o[t] = wmma_bf16(ap, frag_krow(&Vts[16 * t + n][0], h), o[t]);
    }

    // normalize, recombine heads into Z [B,T,D]
    unsigned short* zp = Zb + ((size_t)b * T_ + 16 * qi + 8 * h) * D_ + head * HD_ + n;
#pragma unroll
    for (int v = 0; v < 8; ++v) {
        int off = v * D_;
        float inv = 1.0f / lrow[v];
        zp[off +  0] = f2bfu(o[0][v] * inv);
        zp[off + 16] = f2bfu(o[1][v] * inv);
        zp[off + 32] = f2bfu(o[2][v] * inv);
        zp[off + 48] = f2bfu(o[3][v] * inv);
    }
}

// ---------------------------------------------------------------------------
// Kernel 3: output projection  Z(bf16) @ W_O + b_O -> f32, 16x64 tile / wave.
// grid = (D/64, (B*T)/16), block = 32
// ---------------------------------------------------------------------------
__global__ void out_proj_kernel(const unsigned short* __restrict__ Zb,
                                const unsigned short* __restrict__ WtO,
                                const float* __restrict__ bias,
                                float* __restrict__ Out) {
    __shared__ __align__(16) unsigned short Xs[16][32];
    __shared__ __align__(16) unsigned short Ws[64][32];

    const int lane = threadIdx.x;
    const int n = lane & 15, h = lane >> 4;
    const int ct = blockIdx.x;
    const int ry = blockIdx.y;

    f32x8 acc[4] = {zero8(), zero8(), zero8(), zero8()};

    for (int kk = 0; kk < D_; kk += 32) {
        __syncthreads();
        {
            const unsigned short* xs = Zb + (size_t)(ry * 16 + n) * D_ + kk + 16 * h;
            cp_b128(xs,     &Xs[n][16 * h]);
            cp_b128(xs + 8, &Xs[n][16 * h + 8]);
        }
        {
#pragma unroll
            for (int r = 0; r < 2; ++r) {
                int wr = lane + 32 * r;
                const unsigned short* src = WtO + (size_t)(64 * ct + wr) * D_ + kk;
#pragma unroll
                for (int i = 0; i < 4; ++i) cp_b128(src + 8 * i, &Ws[wr][8 * i]);
            }
        }
        cp_wait();
        __syncthreads();
        bf16x16 a = frag_krow(&Xs[n][0], h);
#pragma unroll
        for (int i = 0; i < 4; ++i)
            acc[i] = wmma_bf16(a, frag_krow(&Ws[16 * i + n][0], h), acc[i]);
    }

    float* op = Out + (size_t)(ry * 16 + 8 * h) * D_;
#pragma unroll
    for (int i = 0; i < 4; ++i) {
        int col = 64 * ct + 16 * i + n;
        float bv = bias[col];
#pragma unroll
        for (int v = 0; v < 8; ++v) op[v * D_ + col] = acc[i][v] + bv;
    }
}

// ---------------------------------------------------------------------------
extern "C" void kernel_launch(void* const* d_in, const int* in_sizes, int n_in,
                              void* d_out, int out_size, void* d_ws, size_t ws_size,
                              hipStream_t stream) {
    (void)in_sizes; (void)n_in; (void)out_size; (void)ws_size;

    const float* X  = (const float*)d_in[0];
    const float* WK = (const float*)d_in[1];
    const float* bK = (const float*)d_in[2];
    const float* WQ = (const float*)d_in[3];
    const float* bQ = (const float*)d_in[4];
    const float* WV = (const float*)d_in[5];
    const float* bV = (const float*)d_in[6];
    const float* WO = (const float*)d_in[7];
    const float* bO = (const float*)d_in[8];

    // workspace layout (bf16 halves); total ~44 MB
    unsigned short* Xb  = (unsigned short*)d_ws;     // [B*T, D]
    unsigned short* Qb  = Xb  + NELEM;               // [B,H,T,hd]
    unsigned short* Kb  = Qb  + NELEM;               // [B,H,T,hd]
    unsigned short* Vt  = Kb  + NELEM;               // [B,H,hd,T]
    unsigned short* Zb  = Vt  + NELEM;               // [B,T,D]
    unsigned short* WtK = Zb  + NELEM;               // [out,in] x4
    unsigned short* WtQ = WtK + WELEM;
    unsigned short* WtV = WtQ + WELEM;
    unsigned short* WtO = WtV + WELEM;

    x_to_bf16_kernel<<<dim3((unsigned)(NELEM / 8 / 256)), dim3(256), 0, stream>>>(X, Xb);
    w_transpose_kernel<<<dim3(16, 16, 4), dim3(32, 8), 0, stream>>>(
        WK, WQ, WV, WO, WtK, WtQ, WtV, WtO);

    qkv_proj_kernel<<<dim3(D_ / 64, (B_ * T_) / 16, 3), dim3(32), 0, stream>>>(
        Xb, WtK, bK, WtQ, bQ, WtV, bV, Qb, Kb, Vt);

    flash_attn_kernel<<<dim3(T_ / 16, B_ * H_), dim3(32), 0, stream>>>(Qb, Kb, Vt, Zb);

    out_proj_kernel<<<dim3(D_ / 64, (B_ * T_) / 16), dim3(32), 0, stream>>>(
        Zb, WtO, bO, (float*)d_out);
}
